// MHGAT_51728586113599
// MI455X (gfx1250) — matrix operand order repaired and checked by reading, not compiled, not run on hardware
//
// MHGAT forward for MI455X (gfx1250, wave32, WMMA).
// Heavy GEMMs via v_wmma_f32_16x16x32_f16. Each wave computes a 64x16 output
// strip: 4 WMMAs per K-step sharing one B-fragment. The FLOP-dominant 36
// attention-message GEMMs ([2048,2048]@[2048,128]) run a pure-f16 path:
// softmax probabilities are stored as f16 (A side, aligned v8h loads) and H is
// pre-packed into WMMA B-fragment order (one contiguous 32B v16h load).
// Workspace requirement: ~500 MB (linear layout, no aliasing).

#include <hip/hip_runtime.h>
#include <hip/hip_bf16.h>
#include <math.h>

static constexpr int kN  = 200000;   // NODE_LEN
static constexpr int kE  = 1000000;  // edges
static constexpr int kNT = 2048;
static constexpr int kIN = 256;
static constexpr int kH  = 128;
static constexpr int kL1 = 13;
static constexpr int kL2 = 30;

typedef __attribute__((ext_vector_type(16))) _Float16 v16h;
typedef __attribute__((ext_vector_type(8)))  _Float16 v8h;
typedef __attribute__((ext_vector_type(8)))  float    v8f;

// Fragment layouts used here (contiguous k order, same in A and B):
//   A 16x32 f16 : lane m=lane&15, half=lane>>4; elem e -> K = k0+half*16+e
//   B 32x16 f16 : lane n=lane&15, half=lane>>4; elem e -> K = k0+half*16+e
//   C/D 16x16 f32: lane n=lane&15; vgpr r -> row M = 8*half + r

// Load A fragment from f32 row-major (16B-aligned rows: lda multiple of 4,
// K-offsets multiple of 8) with two float4 loads per 8-run.
__device__ __forceinline__ v16h load_a_f32(const float* __restrict__ A, size_t row,
                                           int lda, int ka0, int ka1, int dorelu) {
  const float4* p0 = (const float4*)(A + row * (size_t)lda + ka0);
  const float4* p1 = (const float4*)(A + row * (size_t)lda + ka1);
  float4 a = p0[0], b = p0[1], c = p1[0], d = p1[1];
  float v[16] = {a.x, a.y, a.z, a.w, b.x, b.y, b.z, b.w,
                 c.x, c.y, c.z, c.w, d.x, d.y, d.z, d.w};
  v16h f;
#pragma unroll
  for (int j = 0; j < 16; ++j) {
    float t = v[j];
    if (dorelu) t = t > 0.f ? t : 0.f;
    f[j] = (_Float16)t;
  }
  return f;
}

// ---------------------------------------------------------------------------
// Generic WMMA GEMM: C[M,N](ldc) = act( A[M,K](lda) @ B[K,N](ldb) + bias[n] )
// One wave per 64x16 output strip; K stepped by 32. MODE 0: plain A.
// MODE 1 (GCN2): A(m,k) = relu( h1[ k<128 ? m : root(m) ][k&127] ),
//                root(m) = rootindex[batch[m]]  (fuses concat+gather+relu).
// ---------------------------------------------------------------------------
template <int MODE>
__global__ __launch_bounds__(256) void wmma_gemm(
    const float* __restrict__ A, int lda,
    const float* __restrict__ B, int ldb,
    float* __restrict__ C, int ldc,
    int M, int N, int K,
    const float* __restrict__ bias, int act,
    const int* __restrict__ batch, const int* __restrict__ rootindex) {
  int wave = blockIdx.x * (blockDim.x >> 5) + (threadIdx.x >> 5);
  int tiles_n = N >> 4;
  int strips_m = M >> 6;  // 64-row strips
  if (wave >= strips_m * tiles_n) return;
  int tm = wave / tiles_n, tn = wave % tiles_n;
  int lane = threadIdx.x & 31;
  int half = lane >> 4, l16 = lane & 15;
  int bcol = tn * 16 + l16;

  int r0 = tm * 64 + l16, r1 = r0 + 16, r2 = r0 + 32, r3 = r0 + 48;
  int rt0 = 0, rt1 = 0, rt2 = 0, rt3 = 0;
  if (MODE == 1) {
    rt0 = rootindex[batch[r0]]; rt1 = rootindex[batch[r1]];
    rt2 = rootindex[batch[r2]]; rt3 = rootindex[batch[r3]];
  }

  v8f acc0 = {}, acc1 = {}, acc2 = {}, acc3 = {};
  for (int k0 = 0; k0 < K; k0 += 32) {
    // shared B fragment for the whole 64-row strip
    v16h bf;
    const float* Bp = B + (size_t)(k0 + half * 16) * ldb + bcol;
#pragma unroll
    for (int e = 0; e < 16; ++e) bf[e] = (_Float16)Bp[(size_t)e * ldb];
    if (k0 + 32 < K) __builtin_prefetch(Bp + (size_t)32 * ldb, 0, 3);

    int ka0 = k0 + half * 16;
    int ka1 = k0 + half * 16 + 8;
    v16h a0, a1, a2, a3;
    if (MODE == 0) {
      a0 = load_a_f32(A, (size_t)r0, lda, ka0, ka1, 0);
      a1 = load_a_f32(A, (size_t)r1, lda, ka0, ka1, 0);
      a2 = load_a_f32(A, (size_t)r2, lda, ka0, ka1, 0);
      a3 = load_a_f32(A, (size_t)r3, lda, ka0, ka1, 0);
      if (k0 + 32 < K) __builtin_prefetch(A + (size_t)r0 * lda + k0 + 32, 0, 3);
    } else {
      // runs of 8 never straddle the 128 boundary (k0 multiple of 32)
      int m0 = (ka0 & (kH - 1)), m1 = (ka1 & (kH - 1));
      size_t q00 = (size_t)(ka0 < kH ? r0 : rt0), q01 = (size_t)(ka1 < kH ? r0 : rt0);
      size_t q10 = (size_t)(ka0 < kH ? r1 : rt1), q11 = (size_t)(ka1 < kH ? r1 : rt1);
      size_t q20 = (size_t)(ka0 < kH ? r2 : rt2), q21 = (size_t)(ka1 < kH ? r2 : rt2);
      size_t q30 = (size_t)(ka0 < kH ? r3 : rt3), q31 = (size_t)(ka1 < kH ? r3 : rt3);
      // reuse loader with per-half rows by two half-fragment loads
      v16h t;
      t = load_a_f32(A, q00, kH, m0, m0, 1); a0 = t;
      t = load_a_f32(A, q01, kH, m1, m1, 1);
#pragma unroll
      for (int j = 0; j < 8; ++j) a0[8 + j] = t[j];
      t = load_a_f32(A, q10, kH, m0, m0, 1); a1 = t;
      t = load_a_f32(A, q11, kH, m1, m1, 1);
#pragma unroll
      for (int j = 0; j < 8; ++j) a1[8 + j] = t[j];
      t = load_a_f32(A, q20, kH, m0, m0, 1); a2 = t;
      t = load_a_f32(A, q21, kH, m1, m1, 1);
#pragma unroll
      for (int j = 0; j < 8; ++j) a2[8 + j] = t[j];
      t = load_a_f32(A, q30, kH, m0, m0, 1); a3 = t;
      t = load_a_f32(A, q31, kH, m1, m1, 1);
#pragma unroll
      for (int j = 0; j < 8; ++j) a3[8 + j] = t[j];
    }
    acc0 = __builtin_amdgcn_wmma_f32_16x16x32_f16(false, a0, false, bf, (short)0, acc0, false, false);
    acc1 = __builtin_amdgcn_wmma_f32_16x16x32_f16(false, a1, false, bf, (short)0, acc1, false, false);
    acc2 = __builtin_amdgcn_wmma_f32_16x16x32_f16(false, a2, false, bf, (short)0, acc2, false, false);
    acc3 = __builtin_amdgcn_wmma_f32_16x16x32_f16(false, a3, false, bf, (short)0, acc3, false, false);
  }
  float bv = bias ? bias[bcol] : 0.f;
#pragma unroll
  for (int r = 0; r < 8; ++r) {
    float v0 = acc0[r] + bv, v1 = acc1[r] + bv, v2 = acc2[r] + bv, v3 = acc3[r] + bv;
    if (act) {
      v0 = fmaxf(v0, 0.f); v1 = fmaxf(v1, 0.f);
      v2 = fmaxf(v2, 0.f); v3 = fmaxf(v3, 0.f);
    }
    size_t mrow = (size_t)(tm * 64 + half * 8 + r);
    C[(mrow)      * ldc + bcol] = v0;
    C[(mrow + 16) * ldc + bcol] = v1;
    C[(mrow + 32) * ldc + bcol] = v2;
    C[(mrow + 48) * ldc + bcol] = v3;
  }
}

// ---------------------------------------------------------------------------
// Pure-f16 WMMA GEMM: A is f16 row-major (lda mult of 8), B pre-packed in
// fragment order: pk[ ((kb*2+half)*N + n)*16 + e ] = B[kb*32+half*16+e, n].
// ---------------------------------------------------------------------------
__global__ __launch_bounds__(256) void wmma_gemm_h16(
    const _Float16* __restrict__ A, int lda,
    const _Float16* __restrict__ Bpk,
    float* __restrict__ C, int ldc,
    int M, int N, int K) {
  int wave = blockIdx.x * (blockDim.x >> 5) + (threadIdx.x >> 5);
  int tiles_n = N >> 4;
  int strips_m = M >> 6;
  if (wave >= strips_m * tiles_n) return;
  int tm = wave / tiles_n, tn = wave % tiles_n;
  int lane = threadIdx.x & 31;
  int half = lane >> 4, l16 = lane & 15;
  int bcol = tn * 16 + l16;
  int r0 = tm * 64 + l16, r1 = r0 + 16, r2 = r0 + 32, r3 = r0 + 48;

  v8f acc0 = {}, acc1 = {}, acc2 = {}, acc3 = {};
  for (int k0 = 0; k0 < K; k0 += 32) {
    int kb = k0 >> 5;
    v16h bf = *(const v16h*)(Bpk + (((size_t)kb * 2 + half) * N + bcol) * 16);
    int ka0 = k0 + half * 16;
    int ka1 = k0 + half * 16 + 8;
    v16h a0, a1, a2, a3;
    {
      v8h lo, hi;
      lo = *(const v8h*)(A + (size_t)r0 * lda + ka0);
      hi = *(const v8h*)(A + (size_t)r0 * lda + ka1);
#pragma unroll
      for (int j = 0; j < 8; ++j) { a0[j] = lo[j]; a0[8 + j] = hi[j]; }
      lo = *(const v8h*)(A + (size_t)r1 * lda + ka0);
      hi = *(const v8h*)(A + (size_t)r1 * lda + ka1);
#pragma unroll
      for (int j = 0; j < 8; ++j) { a1[j] = lo[j]; a1[8 + j] = hi[j]; }
      lo = *(const v8h*)(A + (size_t)r2 * lda + ka0);
      hi = *(const v8h*)(A + (size_t)r2 * lda + ka1);
#pragma unroll
      for (int j = 0; j < 8; ++j) { a2[j] = lo[j]; a2[8 + j] = hi[j]; }
      lo = *(const v8h*)(A + (size_t)r3 * lda + ka0);
      hi = *(const v8h*)(A + (size_t)r3 * lda + ka1);
#pragma unroll
      for (int j = 0; j < 8; ++j) { a3[j] = lo[j]; a3[8 + j] = hi[j]; }
    }
    if (k0 + 32 < K) __builtin_prefetch(A + (size_t)r0 * lda + k0 + 32, 0, 3);
    acc0 = __builtin_amdgcn_wmma_f32_16x16x32_f16(false, a0, false, bf, (short)0, acc0, false, false);
    acc1 = __builtin_amdgcn_wmma_f32_16x16x32_f16(false, a1, false, bf, (short)0, acc1, false, false);
    acc2 = __builtin_amdgcn_wmma_f32_16x16x32_f16(false, a2, false, bf, (short)0, acc2, false, false);
    acc3 = __builtin_amdgcn_wmma_f32_16x16x32_f16(false, a3, false, bf, (short)0, acc3, false, false);
  }
#pragma unroll
  for (int r = 0; r < 8; ++r) {
    size_t mrow = (size_t)(tm * 64 + half * 8 + r);
    C[(mrow)      * ldc + bcol] = acc0[r];
    C[(mrow + 16) * ldc + bcol] = acc1[r];
    C[(mrow + 32) * ldc + bcol] = acc2[r];
    C[(mrow + 48) * ldc + bcol] = acc3[r];
  }
}

// Pack f32 [K,N] into f16 WMMA B-fragment order (blocked by kb, so a slice
// starting at k = t*2048 lives at pk + t*2048*N).
__global__ void pack_b_f16(const float* __restrict__ B, _Float16* __restrict__ pk,
                           int K, int N) {
  int g = blockIdx.x * blockDim.x + threadIdx.x;
  if (g >= K * N) return;
  int k = g / N, n = g % N;
  int kb = k >> 5, r = k & 31, half = r >> 4, e = r & 15;
  pk[(((size_t)kb * 2 + half) * N + n) * 16 + e] = (_Float16)B[(size_t)k * N + n];
}

// ---------------------------------------------------------------------------
__global__ void fill_f32(float* p, size_t n, float v) {
  size_t i = (size_t)blockIdx.x * blockDim.x + threadIdx.x;
  size_t st = (size_t)gridDim.x * blockDim.x;
  for (; i < n; i += st) p[i] = v;
}

__global__ void deg_kernel(const int* __restrict__ ecol, float* __restrict__ deg) {
  int g = blockIdx.x * blockDim.x + threadIdx.x;
  if (g < kE) atomicAdd(&deg[ecol[g]], 1.f);
  else if (g < kE + kN) atomicAdd(&deg[g - kE], 1.f);
}

__global__ void dinv_kernel(const float* __restrict__ deg, float* __restrict__ dinv) {
  int i = blockIdx.x * blockDim.x + threadIdx.x;
  if (i < kN) dinv[i] = rsqrtf(fmaxf(deg[i], 1.f));
}

// out[col] += t[row] * dinv[row]*dinv[col]  over real edges + self-loops
__global__ void gcn_scatter(const float* __restrict__ t, const int* __restrict__ erow,
                            const int* __restrict__ ecol, const float* __restrict__ dinv,
                            float* __restrict__ outp) {
  size_t g = (size_t)blockIdx.x * blockDim.x + threadIdx.x;
  if (g >= (size_t)(kE + kN) * 32) return;
  int edge = (int)(g >> 5), lane = (int)(g & 31);
  int r, c;
  if (edge < kE) { r = erow[edge]; c = ecol[edge]; }
  else { r = c = edge - kE; }
  float nrm = dinv[r] * dinv[c];
  const float* srow = t + (size_t)r * kH;
  float* drow = outp + (size_t)c * kH;
  for (int j = lane; j < kH; j += 32) atomicAdd(&drow[j], srow[j] * nrm);
}

__global__ void bias_act(float* __restrict__ h, const float* __restrict__ b,
                         size_t rows, int relu) {
  size_t g = (size_t)blockIdx.x * blockDim.x + threadIdx.x;
  if (g >= rows * kH) return;
  float v = h[g] + b[g & (kH - 1)];
  if (relu) v = fmaxf(v, 0.f);
  h[g] = v;
}

__global__ void gather_rows(const float* __restrict__ src, const int* __restrict__ idx,
                            float* __restrict__ dst, int nrows) {
  size_t g = (size_t)blockIdx.x * blockDim.x + threadIdx.x;
  if (g >= (size_t)nrows * kH) return;
  int r = (int)(g >> 7), d = (int)(g & (kH - 1));
  dst[g] = src[(size_t)idx[r] * kH + d];
}

// dst row 2n = src[parent[n]], row 2n+1 = src[root[n]]  (matches stack([pr, rt], 1))
__global__ void gather_ids(const float* __restrict__ src, const int* __restrict__ par,
                           const int* __restrict__ root, float* __restrict__ dst) {
  size_t g = (size_t)blockIdx.x * blockDim.x + threadIdx.x;
  if (g >= (size_t)2 * kNT * kH) return;
  int r = (int)(g >> 7), d = (int)(g & (kH - 1));
  int n = r >> 1;
  int idx = (r & 1) ? root[n] : par[n];
  dst[g] = src[(size_t)idx * kH + d];
}

// Additive attention + sigmoid fuse. qW [NT,128]=qv@Wq, kW [NT*L,128]=kv@Wk.
__global__ void addatt_fuse(const float* __restrict__ qW, const float* __restrict__ kW,
                            const float* __restrict__ kv, const float* __restrict__ qv,
                            const float* __restrict__ vvec, const float* __restrict__ fw,
                            const float* __restrict__ fb, float* __restrict__ outp, int L) {
  int n = blockIdx.x, d = threadIdx.x;  // 128 threads
  __shared__ float red[128];
  __shared__ float sc[32];
  __shared__ float alpha_sh;
  float qd = qW[(size_t)n * kH + d];
  float vd = vvec[d];
  for (int l = 0; l < L; ++l) {
    red[d] = tanhf(qd + kW[((size_t)n * L + l) * kH + d]) * vd;
    __syncthreads();
    for (int s = 64; s > 0; s >>= 1) { if (d < s) red[d] += red[d + s]; __syncthreads(); }
    if (d == 0) sc[l] = red[0];
    __syncthreads();
  }
  float mx = -1e30f;
  for (int l = 0; l < L; ++l) mx = fmaxf(mx, sc[l]);
  float ssum = 0.f;
  for (int l = 0; l < L; ++l) ssum += expf(sc[l] - mx);
  float att = 0.f;
  for (int l = 0; l < L; ++l)
    att += (expf(sc[l] - mx) / ssum) * kv[((size_t)n * L + l) * kH + d];
  float qvd = qv[(size_t)n * kH + d];
  red[d] = qvd * fw[d] + att * fw[kH + d];
  __syncthreads();
  for (int s = 64; s > 0; s >>= 1) { if (d < s) red[d] += red[d + s]; __syncthreads(); }
  if (d == 0) alpha_sh = 1.f / (1.f + expf(-(red[0] + fb[0])));
  __syncthreads();
  float a = alpha_sh;
  outp[(size_t)n * kH + d] = a * qvd + (1.f - a) * att;
}

// Per-node multi-head attention core: o = softmax(qk^T/4) v, 8 heads of dim 16.
__global__ void mha_attn(const float* __restrict__ q, const float* __restrict__ k,
                         const float* __restrict__ v, float* __restrict__ o, int L) {
  int n = blockIdx.x, tid = threadIdx.x;  // 256 threads
  __shared__ float ks[kL2 * kH];
  __shared__ float vs[kL2 * kH];
  size_t base = (size_t)n * L * kH;
  for (int i = tid; i < L * kH; i += 256) { ks[i] = k[base + i]; vs[i] = v[base + i]; }
  __syncthreads();
  if (tid < 8 * L) {
    int l = tid >> 3, h = tid & 7;
    float qr[16];
#pragma unroll
    for (int d = 0; d < 16; ++d) qr[d] = q[base + (size_t)l * kH + h * 16 + d];
    float sc[kL2];
    float mx = -1e30f;
    for (int m = 0; m < L; ++m) {
      float s = 0.f;
#pragma unroll
      for (int d = 0; d < 16; ++d) s += qr[d] * ks[m * kH + h * 16 + d];
      s *= 0.25f;  // 1/sqrt(16)
      sc[m] = s;
      mx = fmaxf(mx, s);
    }
    float sum = 0.f;
    for (int m = 0; m < L; ++m) { sc[m] = expf(sc[m] - mx); sum += sc[m]; }
    float inv = 1.f / sum;
#pragma unroll
    for (int d = 0; d < 16; ++d) {
      float a = 0.f;
      for (int m = 0; m < L; ++m) a += sc[m] * vs[m * kH + h * 16 + d];
      o[base + (size_t)l * kH + h * 16 + d] = a * inv;
    }
  }
}

// src[p][n] = H[t1][n].a_src[t1,t2] ; dst[p][n] = H[t2][n].a_dst[t1,t2]
__global__ void gat_srcdst(const float* __restrict__ H, const float* __restrict__ asrc,
                           const float* __restrict__ adst, float* __restrict__ sd) {
  int g = blockIdx.x * blockDim.x + threadIdx.x;
  if (g >= 2 * 36 * kNT) return;
  int which = g / (36 * kNT);
  int rem = g % (36 * kNT);
  int p = rem / kNT, n = rem % kNT;
  int t1 = p / 6, t2 = p % 6;
  const float* vec = (which == 0 ? asrc : adst) + (size_t)p * kH;
  const float* Hr = H + ((size_t)(which == 0 ? t1 : t2) * kNT + n) * kH;
  float acc = 0.f;
  for (int d = 0; d < kH; ++d) acc += Hr[d] * vec[d];
  sd[g] = acc;
}

// Row-wise masked leaky-relu softmax, emitted directly as f16 (WMMA A operand):
// P[n,m] = softmax_m( adj>0 ? lrelu(src[n]+dst[m]) : -1e9 )
__global__ void gat_softmax(const float* __restrict__ adjp, const float* __restrict__ srcp,
                            const float* __restrict__ dstp, _Float16* __restrict__ P) {
  int n = blockIdx.x, tid = threadIdx.x;  // 256 threads
  __shared__ float sh[kNT];
  __shared__ float red[256];
  float sn = srcp[n];
  const float* arow = adjp + (size_t)n * kNT;
  float lmax = -1e30f;
  for (int m = tid; m < kNT; m += 256) {
    float e = sn + dstp[m];
    float s = (arow[m] > 0.f) ? (e > 0.f ? e : 0.1f * e) : -1e9f;
    sh[m] = s;
    lmax = fmaxf(lmax, s);
  }
  red[tid] = lmax; __syncthreads();
  for (int s = 128; s > 0; s >>= 1) { if (tid < s) red[tid] = fmaxf(red[tid], red[tid + s]); __syncthreads(); }
  float gmax = red[0]; __syncthreads();
  float lsum = 0.f;
  for (int m = tid; m < kNT; m += 256) lsum += expf(sh[m] - gmax);
  red[tid] = lsum; __syncthreads();
  for (int s = 128; s > 0; s >>= 1) { if (tid < s) red[tid] += red[tid + s]; __syncthreads(); }
  float ginv = 1.f / red[0];
  for (int m = tid; m < kNT; m += 256)
    P[(size_t)n * kNT + m] = (_Float16)(expf(sh[m] - gmax) * ginv);
}

// Type-level attention: tw [NT*6,128] = xs@W+b ; s_t = tanh(tw).q ; out = sum softmax(s)_t xs_t
__global__ void type_att_reduce(const float* __restrict__ tw, const float* __restrict__ xs,
                                const float* __restrict__ q, float* __restrict__ outp, int relu) {
  int n = blockIdx.x, d = threadIdx.x;  // 128 threads
  __shared__ float red[128];
  __shared__ float sc[6];
  float qd = q[d];
  for (int t = 0; t < 6; ++t) {
    red[d] = tanhf(tw[((size_t)n * 6 + t) * kH + d]) * qd;
    __syncthreads();
    for (int s = 64; s > 0; s >>= 1) { if (d < s) red[d] += red[d + s]; __syncthreads(); }
    if (d == 0) sc[t] = red[0];
    __syncthreads();
  }
  float mx = -1e30f;
  for (int t = 0; t < 6; ++t) mx = fmaxf(mx, sc[t]);
  float ssum = 0.f;
  for (int t = 0; t < 6; ++t) ssum += expf(sc[t] - mx);
  float acc = 0.f;
  for (int t = 0; t < 6; ++t)
    acc += (expf(sc[t] - mx) / ssum) * xs[((size_t)n * 6 + t) * kH + d];
  if (relu) acc = fmaxf(acc, 0.f);
  outp[(size_t)n * kH + d] = acc;
}

__global__ void final_head(const float* __restrict__ x, const float* __restrict__ W,
                           const float* __restrict__ b, float* __restrict__ outp) {
  int n = blockIdx.x * blockDim.x + threadIdx.x;
  if (n >= kNT) return;
  float z0 = b[0], z1 = b[1], z2 = b[2], z3 = b[3];
  const float* xr = x + (size_t)n * kH;
  for (int d = 0; d < kH; ++d) {
    float xv = xr[d];
    z0 += xv * W[d * 4 + 0]; z1 += xv * W[d * 4 + 1];
    z2 += xv * W[d * 4 + 2]; z3 += xv * W[d * 4 + 3];
  }
  float mx = fmaxf(fmaxf(z0, z1), fmaxf(z2, z3));
  float s = expf(z0 - mx) + expf(z1 - mx) + expf(z2 - mx) + expf(z3 - mx);
  float ls = logf(s) + mx;
  outp[n * 4 + 0] = z0 - ls; outp[n * 4 + 1] = z1 - ls;
  outp[n * 4 + 2] = z2 - ls; outp[n * 4 + 3] = z3 - ls;
}

// ---------------------------------------------------------------------------
extern "C" void kernel_launch(void* const* d_in, const int* in_sizes, int n_in,
                              void* d_out, int out_size, void* d_ws, size_t ws_size,
                              hipStream_t stream) {
  (void)in_sizes; (void)n_in; (void)out_size; (void)ws_size;
  const float* x      = (const float*)d_in[0];
  const int*   eidx   = (const int*)  d_in[1];
  const int*   batch  = (const int*)  d_in[2];
  const int*   rootix = (const int*)  d_in[3];
  const int*   tx0    = (const int*)  d_in[4];
  const int*   tx1    = (const int*)  d_in[5];
  const int*   tx2    = (const int*)  d_in[6];
  const int*   root1  = (const int*)  d_in[7];
  const int*   par1   = (const int*)  d_in[8];
  const int*   root2  = (const int*)  d_in[9];
  const int*   par2   = (const int*)  d_in[10];
  const int*   tx3    = (const int*)  d_in[11];
  const int*   root3  = (const int*)  d_in[12];
  const int*   tx4    = (const int*)  d_in[13];
  const int*   root4  = (const int*)  d_in[14];
  const float* adj    = (const float*)d_in[15];
  const float* W1     = (const float*)d_in[16];
  const float* b1     = (const float*)d_in[17];
  const float* W2     = (const float*)d_in[18];
  const float* b2     = (const float*)d_in[19];
  const float* mhaW   = (const float*)d_in[20];
  const float* mhaB   = (const float*)d_in[21];
  const float* attWq  = (const float*)d_in[22];
  const float* attWk  = (const float*)d_in[23];
  const float* attV   = (const float*)d_in[24];
  const float* fuseW  = (const float*)d_in[25];
  const float* fuseB  = (const float*)d_in[26];
  const float* gatW   = (const float*)d_in[27];
  const float* gatAs  = (const float*)d_in[28];
  const float* gatAd  = (const float*)d_in[29];
  const float* taW    = (const float*)d_in[30];
  const float* taB    = (const float*)d_in[31];
  const float* taQ    = (const float*)d_in[32];
  const float* gc2W   = (const float*)d_in[33];
  const float* gc2b   = (const float*)d_in[34];
  const float* finW   = (const float*)d_in[35];
  const float* finB   = (const float*)d_in[36];
  float* out = (float*)d_out;
  float* ws  = (float*)d_ws;

  // ---- workspace layout (floats; all sizes multiples of 8 -> 32B aligned) ----
  size_t o = 0;
  auto alloc = [&](size_t n) { size_t r = o; o += n; return r; };
  float* deg  = ws + alloc(kN);
  float* dinv = ws + alloc(kN);
  float* txw  = ws + alloc((size_t)kN * kH);   // GEMM output pre-aggregation
  float* h1   = ws + alloc((size_t)kN * kH);
  float* h2   = ws + alloc((size_t)kN * kH);
  float* x0b  = ws + alloc((size_t)6 * kNT * kH);
  float* Hb   = ws + alloc((size_t)6 * kNT * kH);
  float* sd   = ws + alloc((size_t)2 * 36 * kNT);
  _Float16* Pb  = (_Float16*)(ws + alloc((size_t)kNT * kNT / 2));   // f16 probs
  _Float16* Hpk = (_Float16*)(ws + alloc((size_t)6 * kNT * kH / 2)); // packed H
  float* mbuf = ws + alloc((size_t)kNT * 6 * kH);
  float* tab  = ws + alloc((size_t)kNT * 6 * kH);
  float* X1   = ws + alloc((size_t)6 * kNT * kH);
  float* hx   = ws + alloc((size_t)6 * kNT * kH);
  float* m5   = ws + alloc((size_t)kNT * 6 * kH);
  float* x25  = ws + alloc((size_t)kNT * kH);
  float* glst = ws + alloc((size_t)kNT * kH);
  float* gids = ws + alloc((size_t)2 * kNT * kH);
  float* gqw  = ws + alloc((size_t)kNT * kH);
  float* gkw  = ws + alloc((size_t)2 * kNT * kH);
  float* mA   = ws + alloc((size_t)kNT * kL2 * kH);
  float* mB   = ws + alloc((size_t)kNT * kL2 * kH);
  float* mC   = ws + alloc((size_t)kNT * kL2 * kH);
  float* mD   = ws + alloc((size_t)kNT * kL2 * kH);

  const int* erow = eidx;
  const int* ecol = eidx + kE;

  auto gemm = [&](const float* A, int lda, const float* B, int ldb, float* C, int ldc,
                  int M, int N, int K, const float* bias, int act) {
    int tiles = (M / 64) * (N / 16);
    int blocks = (tiles + 7) / 8;
    wmma_gemm<0><<<blocks, 256, 0, stream>>>(A, lda, B, ldb, C, ldc, M, N, K, bias, act,
                                             (const int*)nullptr, (const int*)nullptr);
  };
  auto gemm_h16 = [&](const _Float16* A, int lda, const _Float16* Bpk, float* C, int ldc,
                      int M, int N, int K) {
    int tiles = (M / 64) * (N / 16);
    int blocks = (tiles + 7) / 8;
    wmma_gemm_h16<<<blocks, 256, 0, stream>>>(A, lda, Bpk, C, ldc, M, N, K);
  };
  auto gather = [&](const float* src, const int* idx, float* dst, int nrows) {
    size_t total = (size_t)nrows * kH;
    gather_rows<<<(int)((total + 255) / 256), 256, 0, stream>>>(src, idx, dst, nrows);
  };

  // ---- degree / norm ----
  fill_f32<<<2048, 256, 0, stream>>>(deg, (size_t)kN, 0.f);
  fill_f32<<<4096, 256, 0, stream>>>(h1, (size_t)kN * kH, 0.f);
  fill_f32<<<4096, 256, 0, stream>>>(h2, (size_t)kN * kH, 0.f);
  deg_kernel<<<(kE + kN + 255) / 256, 256, 0, stream>>>(ecol, deg);
  dinv_kernel<<<(kN + 255) / 256, 256, 0, stream>>>(deg, dinv);

  // ---- GCN layer 1: h1 = segsum((x@W1)[row]*norm) + b1 ----
  gemm(x, kIN, W1, kH, txw, kH, kN, kH, kIN, nullptr, 0);
  {
    size_t tthr = (size_t)(kE + kN) * 32;
    gcn_scatter<<<(int)((tthr + 255) / 256), 256, 0, stream>>>(txw, erow, ecol, dinv, h1);
  }
  bias_act<<<(int)(((size_t)kN * kH + 255) / 256), 256, 0, stream>>>(h1, b1, kN, 0);

  // ---- GCN layer 2 (fused relu(concat(h1, h1[root])) inside A-loader) ----
  {
    int tiles = (kN / 64) * (kH / 16);
    wmma_gemm<1><<<(tiles + 7) / 8, 256, 0, stream>>>(h1, kH, W2, kH, txw, kH,
                                                      kN, kH, kIN, nullptr, 0, batch, rootix);
    size_t tthr = (size_t)(kE + kN) * 32;
    gcn_scatter<<<(int)((tthr + 255) / 256), 256, 0, stream>>>(txw, erow, ecol, dinv, h2);
  }
  bias_act<<<(int)(((size_t)kN * kH + 255) / 256), 256, 0, stream>>>(h2, b2, kN, 1);

  // ---- x0[0], x0[5] = h2[type_x0] ----
  gather(h2, tx0, x0b + 0 * (size_t)kNT * kH, kNT);
  gather(h2, tx0, x0b + 5 * (size_t)kNT * kH, kNT);

  // ---- x0[1], x0[2]: additive attention over {parent, root} + sigmoid fuse ----
  for (int i = 0; i < 2; ++i) {
    const int* tx = (i == 0) ? tx1 : tx2;
    const int* rt = (i == 0) ? root1 : root2;
    const int* pr = (i == 0) ? par1 : par2;
    gather(h2, tx, glst, kNT);
    gather_ids<<<(int)(((size_t)2 * kNT * kH + 255) / 256), 256, 0, stream>>>(h2, pr, rt, gids);
    gemm(glst, kH, attWq + (size_t)i * kH * kH, kH, gqw, kH, kNT, kH, kH, nullptr, 0);
    gemm(gids, kH, attWk + (size_t)i * kH * kH, kH, gkw, kH, 2 * kNT, kH, kH, nullptr, 0);
    addatt_fuse<<<kNT, 128, 0, stream>>>(gqw, gkw, gids, glst, attV + (size_t)i * kH,
                                         fuseW + (size_t)i * 2 * kH, fuseB + i,
                                         x0b + (size_t)(1 + i) * kNT * kH, 2);
  }

  // ---- x0[3], x0[4]: MHA over sequence + additive attention with root + fuse ----
  for (int i = 0; i < 2; ++i) {
    int L = (i == 0) ? kL1 : kL2;
    const int* seqidx = (i == 0) ? tx3 : tx4;
    const int* rootid = (i == 0) ? root3 : root4;
    int rows = kNT * L;
    gather(h2, seqidx, mA, rows);  // seq in
    for (int j = 0; j < 3; ++j) {
      float* dst = (j == 0) ? mB : (j == 1) ? mC : mD;  // q, k, v
      gemm(mA, kH, mhaW + (size_t)(i * 4 + j) * kH * kH, kH, dst, kH,
           rows, kH, kH, mhaB + (size_t)(i * 4 + j) * kH, 0);
    }
    mha_attn<<<kNT, 256, 0, stream>>>(mB, mC, mD, mA, L);       // a@v -> mA
    gemm(mA, kH, mhaW + (size_t)(i * 4 + 3) * kH * kH, kH, mB, kH,
         rows, kH, kH, mhaB + (size_t)(i * 4 + 3) * kH, 0);     // seq out -> mB
    gather(h2, rootid, glst, kNT);                               // rt
    gemm(glst, kH, attWq + (size_t)(2 + i) * kH * kH, kH, gqw, kH, kNT, kH, kH, nullptr, 0);
    gemm(mB, kH, attWk + (size_t)(2 + i) * kH * kH, kH, mC, kH, rows, kH, kH, nullptr, 0);
    addatt_fuse<<<kNT, 128, 0, stream>>>(gqw, mC, mB, glst, attV + (size_t)(2 + i) * kH,
                                         fuseW + (size_t)(2 + i) * 2 * kH, fuseB + (2 + i),
                                         x0b + (size_t)(3 + i) * kNT * kH, L);
  }

  // ---- H[t] = x0[t] @ gat_W[t] ; pack all H into WMMA B-fragment order ----
  for (int t = 0; t < 6; ++t)
    gemm(x0b + (size_t)t * kNT * kH, kH, gatW + (size_t)t * kH * kH, kH,
         Hb + (size_t)t * kNT * kH, kH, kNT, kH, kH, nullptr, 0);
  pack_b_f16<<<(6 * kNT * kH + 255) / 256, 256, 0, stream>>>(Hb, Hpk, 6 * kNT, kH);

  // ---- heterogeneous GAT: per-pair masked softmax (f16) + f16 message GEMM ----
  gat_srcdst<<<(2 * 36 * kNT + 255) / 256, 256, 0, stream>>>(Hb, gatAs, gatAd, sd);
  for (int t1 = 0; t1 < 6; ++t1) {
    for (int t2 = 0; t2 < 6; ++t2) {
      int p = t1 * 6 + t2;
      gat_softmax<<<kNT, 256, 0, stream>>>(adj + (size_t)p * kNT * kNT,
                                           sd + (size_t)p * kNT,
                                           sd + (size_t)(36 + p) * kNT, Pb);
      gemm_h16(Pb, kNT, Hpk + (size_t)t2 * kNT * kH,
               mbuf + (size_t)t2 * kH, 6 * kH, kNT, kH, kNT);
    }
    gemm(mbuf, kH, taW + (size_t)t1 * kH * kH, kH, tab, kH,
         kNT * 6, kH, kH, taB + (size_t)t1 * kH, 0);
    type_att_reduce<<<kNT, 128, 0, stream>>>(tab, mbuf, taQ + (size_t)t1 * kH,
                                             X1 + (size_t)t1 * kNT * kH, 1);
  }

  // ---- second hop: hx = X1 @ gc2_W ; msgs5[s] = adj[5,s] @ hx[s] + gc2_b ; at2 ----
  gemm(X1, kH, gc2W, kH, hx, kH, 6 * kNT, kH, kH, nullptr, 0);
  for (int s = 0; s < 6; ++s)
    gemm(adj + (size_t)(30 + s) * kNT * kNT, kNT, hx + (size_t)s * kNT * kH, kH,
         m5 + (size_t)s * kH, 6 * kH, kNT, kH, kNT, gc2b, 0);
  gemm(m5, kH, taW + (size_t)11 * kH * kH, kH, tab, kH,
       kNT * 6, kH, kH, taB + (size_t)11 * kH, 0);
  type_att_reduce<<<kNT, 128, 0, stream>>>(tab, m5, taQ + (size_t)11 * kH, x25, 0);

  // ---- head: log_softmax(x25 @ fin_W + fin_b) ----
  final_head<<<(kNT + 255) / 256, 256, 0, stream>>>(x25, finW, finB, out);
}
